// MicrolensingTransformer_11184094839362
// MI455X (gfx1250) — matrix operand order repaired and checked by reading, not compiled, not run on hardware
//
#include <hip/hip_runtime.h>
#include <hip/hip_bf16.h>
#include <math.h>

// Problem constants (from reference)
#define LYR 4
#define BB  8
#define TT  1024
#define DD  256
#define HH  8
#define DKK 32
#define FFD 1024
#define MM  (BB*TT)          // 8192 rows
#define ATT_SCALE 0.17677669529663687f  // 1/sqrt(32)

typedef __attribute__((ext_vector_type(16))) __bf16 v16bf;
typedef __attribute__((ext_vector_type(8)))  float  v8f;
typedef __attribute__((ext_vector_type(4)))  unsigned int u32x4;
typedef __attribute__((ext_vector_type(4)))  float  f32x4;

// 32-byte fragment: 16 bf16 values (A or B operand of wmma 16x16x32)
struct Frag { u32x4 q0, q1; };

static __device__ inline v16bf frag_bf(const Frag& f) {
  return __builtin_bit_cast(v16bf, f);
}

static __device__ inline unsigned short f2bf(float f) {
  unsigned int u = __float_as_uint(f);
  u += 0x7FFFu + ((u >> 16) & 1u);   // round-to-nearest-even
  return (unsigned short)(u >> 16);
}

// CDNA5 async global -> LDS copy of 16 bytes (tracked by ASYNCcnt).
// ldsOff: LDS byte offset (low 32 bits of a generic shared pointer),
// gaddr : 64-bit global byte address.
static __device__ inline void async_copy_b128(unsigned ldsOff, const void* gptr) {
  unsigned long long ga = (unsigned long long)(uintptr_t)gptr;
  asm volatile("global_load_async_to_lds_b128 %0, %1, off"
               :: "v"(ldsOff), "v"(ga) : "memory");
}
static __device__ inline unsigned lds_off(const void* shared_ptr) {
  return (unsigned)(uintptr_t)shared_ptr;   // low 32 bits = LDS byte address
}

// ---------------------------------------------------------------- convert
__global__ __launch_bounds__(256) void cvt_f32_bf16(const float* __restrict__ in,
                                                    unsigned short* __restrict__ out,
                                                    int n) {
  int i = blockIdx.x * 256 + threadIdx.x;
  if (i < n) out[i] = f2bf(in[i]);
}

// ---------------------------------------------------------------- LayerNorm
// One wave per row of x (D=256, 8 floats/lane). DOUBLE=1 applies a second LN
// (q_norm) on the result. Output bf16 [M,D].
template<int DOUBLE>
__global__ __launch_bounds__(128) void ln_kernel(const float* __restrict__ x,
                                                 const float* __restrict__ g1,
                                                 const float* __restrict__ b1,
                                                 const float* __restrict__ g2,
                                                 const float* __restrict__ b2,
                                                 unsigned short* __restrict__ out) {
  int row  = blockIdx.x * 4 + (threadIdx.x >> 5);
  int lane = threadIdx.x & 31;
  const float* xr = x + (size_t)row * DD + lane * 8;
  float v[8];
  f32x4 p0 = *(const f32x4*)(xr);
  f32x4 p1 = *(const f32x4*)(xr + 4);
#pragma unroll
  for (int i = 0; i < 4; ++i) { v[i] = p0[i]; v[4 + i] = p1[i]; }

#pragma unroll
  for (int pass = 0; pass < (DOUBLE ? 2 : 1); ++pass) {
    const float* g = pass ? g2 : g1;
    const float* b = pass ? b2 : b1;
    float s = 0.f;
#pragma unroll
    for (int i = 0; i < 8; ++i) s += v[i];
#pragma unroll
    for (int m = 1; m <= 16; m <<= 1) s += __shfl_xor(s, m, 32);
    float mean = s * (1.f / DD);
    float vs = 0.f;
#pragma unroll
    for (int i = 0; i < 8; ++i) { float d = v[i] - mean; vs += d * d; }
#pragma unroll
    for (int m = 1; m <= 16; m <<= 1) vs += __shfl_xor(vs, m, 32);
    float rs = rsqrtf(vs * (1.f / DD) + 1e-6f);
#pragma unroll
    for (int i = 0; i < 8; ++i)
      v[i] = (v[i] - mean) * rs * g[lane * 8 + i] + b[lane * 8 + i];
  }
  unsigned short o[8];
#pragma unroll
  for (int i = 0; i < 8; ++i) o[i] = f2bf(v[i]);
  *(u32x4*)(out + (size_t)row * DD + lane * 8) = *(const u32x4*)o;
}

// ---------------------------------------------------------------- GEMM
// C[M,N] = epilogue( A[M,K](bf16) @ W[N,K]^T(bf16) + bias )
// Tiles: 128x64 per WG (8 waves, 32x32 per wave), K staged 64 deep through
// double-buffered LDS using CDNA5 async global->LDS loads: stage k+1's loads
// are issued before waiting on stage k (s_wait_asynccnt 6 keeps only the
// newest batch outstanding), overlapping HBM latency with WMMA.
// EPI 0: store f32 (QKV)        -> outF[M,N]
// EPI 1: x += tanh(a)*0.5*val   -> outF = x [M,256] f32 (Wo)
// EPI 2: gelu(val) -> bf16      -> outH[M,N] (W1)
// EPI 3: x = clip(x+tanh(b)*val)-> outF = x [M,256] f32 (W2)
#define LDA 72   // padded LDS row stride (elements) for 64-deep K tiles
template<int EPI>
__global__ __launch_bounds__(256) void gemm_bf16(const unsigned short* __restrict__ A,
                                                 const unsigned short* __restrict__ W,
                                                 const float* __restrict__ bias,
                                                 float* __restrict__ outF,
                                                 unsigned short* __restrict__ outH,
                                                 const float* __restrict__ scal,
                                                 int M, int N, int K) {
  __shared__ __align__(16) unsigned short As[2][128 * LDA];
  __shared__ __align__(16) unsigned short Ws[2][64 * LDA];
  const int tid  = threadIdx.x;
  const int lane = tid & 31;
  const int wv   = tid >> 5;
  const int m0   = blockIdx.y * 128;
  const int n0   = blockIdx.x * 64;
  const int mw   = (wv & 3) * 32;
  const int nw   = (wv >> 2) * 32;
  const int hi   = lane >> 4;
  const int lo   = lane & 15;

  v8f acc[2][2];
  acc[0][0] = (v8f)0.f; acc[0][1] = (v8f)0.f;
  acc[1][0] = (v8f)0.f; acc[1][1] = (v8f)0.f;

  // issue one K-stage (64 deep) of async global->LDS copies: 6 per thread
  auto issue_stage = [&](int buf, int k0) {
#pragma unroll
    for (int i = 0; i < 4; ++i) {          // A tile 128x64: 4 chunks/thread
      int c = tid * 4 + i;
      int r = c >> 3, kc = (c & 7) * 8;
      async_copy_b128(lds_off(&As[buf][r * LDA + kc]),
                      A + (size_t)(m0 + r) * K + k0 + kc);
    }
#pragma unroll
    for (int i = 0; i < 2; ++i) {          // W tile 64x64: 2 chunks/thread
      int c = tid * 2 + i;
      int r = c >> 3, kc = (c & 7) * 8;
      async_copy_b128(lds_off(&Ws[buf][r * LDA + kc]),
                      W + (size_t)(n0 + r) * K + k0 + kc);
    }
  };

  issue_stage(0, 0);
  int cur = 0;
  for (int k0 = 0; k0 < K; k0 += 64) {
    const bool hasNext = (k0 + 64) < K;            // uniform across block
    if (hasNext) {
      issue_stage(cur ^ 1, k0 + 64);
      asm volatile("s_wait_asynccnt 6" ::: "memory");  // current stage resident
    } else {
      asm volatile("s_wait_asynccnt 0" ::: "memory");
    }
    __syncthreads();

#pragma unroll
    for (int ks = 0; ks < 64; ks += 32) {
      Frag a0, a1, b0, b1;
      {
        int r = mw + lo;
        a0.q0 = *(const u32x4*)&As[cur][r * LDA + ks + hi * 16];
        a0.q1 = *(const u32x4*)&As[cur][r * LDA + ks + hi * 16 + 8];
        r = mw + 16 + lo;
        a1.q0 = *(const u32x4*)&As[cur][r * LDA + ks + hi * 16];
        a1.q1 = *(const u32x4*)&As[cur][r * LDA + ks + hi * 16 + 8];
      }
      {
        int r = nw + lo;
        b0.q0 = *(const u32x4*)&Ws[cur][r * LDA + ks + hi * 16];
        b0.q1 = *(const u32x4*)&Ws[cur][r * LDA + ks + hi * 16 + 8];
        r = nw + 16 + lo;
        b1.q0 = *(const u32x4*)&Ws[cur][r * LDA + ks + hi * 16];
        b1.q1 = *(const u32x4*)&Ws[cur][r * LDA + ks + hi * 16 + 8];
      }
      acc[0][0] = __builtin_amdgcn_wmma_f32_16x16x32_bf16(false, frag_bf(a0), false, frag_bf(b0), (short)0, acc[0][0], false, false);
      acc[0][1] = __builtin_amdgcn_wmma_f32_16x16x32_bf16(false, frag_bf(a0), false, frag_bf(b1), (short)0, acc[0][1], false, false);
      acc[1][0] = __builtin_amdgcn_wmma_f32_16x16x32_bf16(false, frag_bf(a1), false, frag_bf(b0), (short)0, acc[1][0], false, false);
      acc[1][1] = __builtin_amdgcn_wmma_f32_16x16x32_bf16(false, frag_bf(a1), false, frag_bf(b1), (short)0, acc[1][1], false, false);
    }
    __syncthreads();   // all waves done reading As[cur]/Ws[cur]
    cur ^= 1;
  }

  float sc = 0.f;
  if (EPI == 1 || EPI == 3) sc = tanhf(scal[0]);

#pragma unroll
  for (int i = 0; i < 2; ++i)
#pragma unroll
    for (int j = 0; j < 2; ++j)
#pragma unroll
      for (int v = 0; v < 8; ++v) {
        int row = m0 + mw + i * 16 + hi * 8 + v;
        int col = n0 + nw + j * 16 + lo;
        float val = acc[i][j][v] + bias[col];
        if (EPI == 0) {
          outF[(size_t)row * N + col] = val;
        } else if (EPI == 1) {
          size_t idx = (size_t)row * DD + col;
          outF[idx] = outF[idx] + sc * 0.5f * val;
        } else if (EPI == 2) {
          float g = 0.5f * val * (1.f + erff(val * 0.70710678118654752f));
          outH[(size_t)row * N + col] = f2bf(g);
        } else {
          size_t idx = (size_t)row * DD + col;
          float xv = outF[idx] + sc * val;
          xv = fminf(fmaxf(xv, -100.f), 100.f);
          outF[idx] = xv;
        }
      }
}

// ---------------------------------------------------------------- QKV split
// qkv [M,768] f32 -> L2-normalized Q,K [BH,T,32] bf16, V transposed [BH,32,T] bf16
__global__ __launch_bounds__(256) void qkv_split(const float* __restrict__ qkv,
                                                 unsigned short* __restrict__ Qb,
                                                 unsigned short* __restrict__ Kb,
                                                 unsigned short* __restrict__ Vtb) {
  int idx = blockIdx.x * 256 + threadIdx.x;
  if (idx >= MM * HH) return;
  int row = idx >> 3;       // b*T + t
  int hd  = idx & 7;
  int b = row >> 10, t = row & (TT - 1);
  const float* q = qkv + (size_t)row * (3 * DD) + hd * DKK;
  const float* k = q + DD;
  const float* v = q + 2 * DD;
  float sq = 0.f, sk = 0.f;
#pragma unroll
  for (int j = 0; j < DKK; ++j) { sq += q[j] * q[j]; sk += k[j] * k[j]; }
  float iq = 1.f / fmaxf(sqrtf(sq), 1e-8f);
  float ik = 1.f / fmaxf(sqrtf(sk), 1e-8f);
  int bh = b * HH + hd;
  size_t qo = ((size_t)bh * TT + t) * DKK;
#pragma unroll
  for (int j = 0; j < DKK; ++j) {
    Qb[qo + j] = f2bf(q[j] * iq);
    Kb[qo + j] = f2bf(k[j] * ik);
    Vtb[((size_t)bh * DKK + j) * TT + t] = f2bf(v[j]);
  }
}

// ---------------------------------------------------------------- Flash attention
// One wave = one (bh, 16-row query tile). Online softmax over 32-col steps.
__global__ __launch_bounds__(128) void attn_kernel(const unsigned short* __restrict__ Qb,
                                                   const unsigned short* __restrict__ Kb,
                                                   const unsigned short* __restrict__ Vtb,
                                                   const unsigned char* __restrict__ pad,
                                                   unsigned short* __restrict__ aob) {
  __shared__ __align__(16) unsigned short Pld[4][16 * 32];
  const int lane = threadIdx.x & 31;
  const int wv   = threadIdx.x >> 5;
  const int hi   = lane >> 4;
  const int lo   = lane & 15;
  int tile   = blockIdx.x * 4 + wv;
  int bh     = tile >> 6;
  int t_base = (tile & 63) * 16;
  int b = bh >> 3, h = bh & 7;

  Frag qa;
  {
    const unsigned short* qr = Qb + ((size_t)bh * TT + (t_base + lo)) * DKK;
    qa.q0 = *(const u32x4*)(qr + hi * 16);
    qa.q1 = *(const u32x4*)(qr + hi * 16 + 8);
  }
  v8f O0 = (v8f)0.f, O1 = (v8f)0.f;
  float mrun[8], lrun[8];
#pragma unroll
  for (int v = 0; v < 8; ++v) { mrun[v] = -1e30f; lrun[v] = 0.f; }

  unsigned short* P = Pld[wv];
  const int t_max = t_base + 15;
  for (int s0 = 0; s0 <= t_max; s0 += 32) {
    Frag kb0, kb1;
    {
      const unsigned short* kr = Kb + ((size_t)bh * TT + (s0 + lo)) * DKK + hi * 16;
      kb0.q0 = *(const u32x4*)(kr);
      kb0.q1 = *(const u32x4*)(kr + 8);
      kr += 16 * DKK;
      kb1.q0 = *(const u32x4*)(kr);
      kb1.q1 = *(const u32x4*)(kr + 8);
    }
    v8f z = (v8f)0.f;
    v8f sc0 = __builtin_amdgcn_wmma_f32_16x16x32_bf16(false, frag_bf(qa), false, frag_bf(kb0), (short)0, z, false, false);
    v8f sc1 = __builtin_amdgcn_wmma_f32_16x16x32_bf16(false, frag_bf(qa), false, frag_bf(kb1), (short)0, z, false, false);

    int sA = s0 + lo, sB = sA + 16;
    bool pA = pad[b * TT + sA] != 0;
    bool pB = pad[b * TT + sB] != 0;
#pragma unroll
    for (int v = 0; v < 8; ++v) {
      int t = t_base + hi * 8 + v;
      float va = fminf(fmaxf(sc0[v] * ATT_SCALE, -10.f), 10.f);
      float vb = fminf(fmaxf(sc1[v] * ATT_SCALE, -10.f), 10.f);
      if (sA > t || pA) va = -10000.f;
      if (sB > t || pB) vb = -10000.f;
      float tm = fmaxf(va, vb);
      tm = fmaxf(tm, __shfl_xor(tm, 1, 32));
      tm = fmaxf(tm, __shfl_xor(tm, 2, 32));
      tm = fmaxf(tm, __shfl_xor(tm, 4, 32));
      tm = fmaxf(tm, __shfl_xor(tm, 8, 32));
      float mnew = fmaxf(mrun[v], tm);
      float corr = __expf(mrun[v] - mnew);
      float pa = __expf(va - mnew);
      float pb = __expf(vb - mnew);
      float ts = pa + pb;
      ts += __shfl_xor(ts, 1, 32);
      ts += __shfl_xor(ts, 2, 32);
      ts += __shfl_xor(ts, 4, 32);
      ts += __shfl_xor(ts, 8, 32);
      lrun[v] = lrun[v] * corr + ts;
      mrun[v] = mnew;
      O0[v] *= corr;
      O1[v] *= corr;
      int r = hi * 8 + v;
      P[r * 32 + lo] = f2bf(pa);
      P[r * 32 + 16 + lo] = f2bf(pb);
    }
    asm volatile("s_wait_dscnt 0" ::: "memory");  // wave-local LDS RAW
    Frag pf, vb0, vb1;
    pf.q0 = *(const u32x4*)&P[lo * 32 + hi * 16];
    pf.q1 = *(const u32x4*)&P[lo * 32 + hi * 16 + 8];
    {
      const unsigned short* vr = Vtb + ((size_t)bh * DKK + lo) * TT + s0 + hi * 16;
      vb0.q0 = *(const u32x4*)(vr);
      vb0.q1 = *(const u32x4*)(vr + 8);
      vr += 16 * TT;
      vb1.q0 = *(const u32x4*)(vr);
      vb1.q1 = *(const u32x4*)(vr + 8);
    }
    O0 = __builtin_amdgcn_wmma_f32_16x16x32_bf16(false, frag_bf(pf), false, frag_bf(vb0), (short)0, O0, false, false);
    O1 = __builtin_amdgcn_wmma_f32_16x16x32_bf16(false, frag_bf(pf), false, frag_bf(vb1), (short)0, O1, false, false);
    asm volatile("" ::: "memory");  // keep next iter's LDS writes after these reads
  }

#pragma unroll
  for (int v = 0; v < 8; ++v) {
    float inv = 1.f / lrun[v];
    int t = t_base + hi * 8 + v;
    size_t base = ((size_t)(b * TT + t)) * DD + h * DKK;
    aob[base + lo] = f2bf(O0[v] * inv);
    aob[base + 16 + lo] = f2bf(O1[v] * inv);
  }
}

// ---------------------------------------------------------------- launch
extern "C" void kernel_launch(void* const* d_in, const int* in_sizes, int n_in,
                              void* d_out, int out_size, void* d_ws, size_t ws_size,
                              hipStream_t stream) {
  const float* x_in  = (const float*)d_in[0];
  const unsigned char* pad = (const unsigned char*)d_in[1];
  const float* ln1_g = (const float*)d_in[2];
  const float* ln1_b = (const float*)d_in[3];
  const float* qn_g  = (const float*)d_in[4];
  const float* qn_b  = (const float*)d_in[5];
  const float* Wqkv  = (const float*)d_in[6];
  const float* bqkv  = (const float*)d_in[7];
  const float* Wo    = (const float*)d_in[8];
  const float* bo    = (const float*)d_in[9];
  const float* ln2_g = (const float*)d_in[10];
  const float* ln2_b = (const float*)d_in[11];
  const float* W1    = (const float*)d_in[12];
  const float* b1    = (const float*)d_in[13];
  const float* W2    = (const float*)d_in[14];
  const float* b2    = (const float*)d_in[15];
  const float* alpha = (const float*)d_in[16];
  const float* beta  = (const float*)d_in[17];
  float* x = (float*)d_out;   // running residual, fp32 [B,T,D]

  char* ws = (char*)d_ws;
  size_t off = 0;
  auto alloc = [&](size_t bytes) -> void* {
    void* p = ws + off;
    off += (bytes + 255) & ~(size_t)255;
    return p;
  };
  unsigned short* xn    = (unsigned short*)alloc((size_t)MM * DD * 2);        // LN out bf16
  float*          qkv   = (float*)         alloc((size_t)MM * 3 * DD * 4);    // raw qkv f32
  unsigned short* Qb    = (unsigned short*)alloc((size_t)MM * DD * 2);
  unsigned short* Kb    = (unsigned short*)alloc((size_t)MM * DD * 2);
  unsigned short* Vtb   = (unsigned short*)alloc((size_t)MM * DD * 2);
  unsigned short* aob   = (unsigned short*)alloc((size_t)MM * DD * 2);        // attn out bf16
  unsigned short* fb    = (unsigned short*)alloc((size_t)MM * FFD * 2);       // FFN hidden bf16
  unsigned short* Wqkvb = (unsigned short*)alloc((size_t)LYR * 3 * DD * DD * 2);
  unsigned short* Wob   = (unsigned short*)alloc((size_t)LYR * DD * DD * 2);
  unsigned short* W1b   = (unsigned short*)alloc((size_t)LYR * FFD * DD * 2);
  unsigned short* W2b   = (unsigned short*)alloc((size_t)LYR * DD * FFD * 2);

  hipMemcpyAsync(x, x_in, (size_t)MM * DD * sizeof(float), hipMemcpyDeviceToDevice, stream);

  { int n = LYR * 3 * DD * DD; cvt_f32_bf16<<<(n + 255) / 256, 256, 0, stream>>>(Wqkv, Wqkvb, n); }
  { int n = LYR * DD * DD;     cvt_f32_bf16<<<(n + 255) / 256, 256, 0, stream>>>(Wo,   Wob,   n); }
  { int n = LYR * FFD * DD;    cvt_f32_bf16<<<(n + 255) / 256, 256, 0, stream>>>(W1,   W1b,   n); }
  { int n = LYR * DD * FFD;    cvt_f32_bf16<<<(n + 255) / 256, 256, 0, stream>>>(W2,   W2b,   n); }

  for (int l = 0; l < LYR; ++l) {
    // ln1 + q_norm -> xn
    ln_kernel<1><<<MM / 4, 128, 0, stream>>>(x, ln1_g + l * DD, ln1_b + l * DD,
                                             qn_g + l * DD, qn_b + l * DD, xn);
    // QKV projection
    gemm_bf16<0><<<dim3((3 * DD) / 64, MM / 128), 256, 0, stream>>>(
        xn, Wqkvb + (size_t)l * 3 * DD * DD, bqkv + l * 3 * DD,
        qkv, nullptr, nullptr, MM, 3 * DD, DD);
    // split + l2norm + V transpose
    qkv_split<<<(MM * HH + 255) / 256, 256, 0, stream>>>(qkv, Qb, Kb, Vtb);
    // flash attention
    attn_kernel<<<(BB * HH * (TT / 16)) / 4, 128, 0, stream>>>(Qb, Kb, Vtb, pad, aob);
    // out projection + residual: x += tanh(alpha)*0.5*(ao@Wo^T+bo)
    gemm_bf16<1><<<dim3(DD / 64, MM / 128), 256, 0, stream>>>(
        aob, Wob + (size_t)l * DD * DD, bo + l * DD,
        x, nullptr, alpha + l, MM, DD, DD);
    // ln2 -> xn
    ln_kernel<0><<<MM / 4, 128, 0, stream>>>(x, ln2_g + l * DD, ln2_b + l * DD,
                                             nullptr, nullptr, xn);
    // FFN1 + gelu -> fb
    gemm_bf16<2><<<dim3(FFD / 64, MM / 128), 256, 0, stream>>>(
        xn, W1b + (size_t)l * FFD * DD, b1 + l * FFD,
        nullptr, fb, nullptr, MM, FFD, DD);
    // FFN2 + residual + clip -> x
    gemm_bf16<3><<<dim3(DD / 64, MM / 128), 256, 0, stream>>>(
        fb, W2b + (size_t)l * DD * FFD, b2 + l * DD,
        x, nullptr, beta + l, MM, DD, FFD);
  }
}
